// DecoderAttentionRotary_13365938225172
// MI455X (gfx1250) — matrix operand order repaired and checked by reading, not compiled, not run on hardware
//
#include <hip/hip_runtime.h>
#include <hip/hip_bf16.h>
#include <stddef.h>

// ---------------- problem constants ----------------
#define B_   2
#define L_   2048
#define D_   2048
#define H_   16
#define HD_  128
#define R_   32
#define ND_  (3 * D_)            // 6144
#define SCALE_ 0.08838834764831845f   // HD^-0.5
#define ELEMS_ (B_ * H_ * L_ * HD_)   // 8388608 (== B*L*D)

// ---------------- WMMA types / helpers ----------------
typedef __attribute__((ext_vector_type(16))) __bf16 v16bf;
typedef __attribute__((ext_vector_type(8)))  float  v8f;
typedef __attribute__((ext_vector_type(4)))  unsigned int u32x4;
typedef __attribute__((ext_vector_type(8)))  int  i32x8;
typedef __attribute__((ext_vector_type(4)))  int  i32x4;

union ABFrag { unsigned int u[8]; v16bf v; };

__device__ inline v8f vzero8() {
    v8f z = {0.f, 0.f, 0.f, 0.f, 0.f, 0.f, 0.f, 0.f};
    return z;
}

__device__ inline v8f wmma_bf16(const ABFrag& a, const ABFrag& b, v8f c) {
    // D = A(16x32 bf16) * B(32x16 bf16) + C(16x16 f32)
    return __builtin_amdgcn_wmma_f32_16x16x32_bf16(
        /*neg_a=*/false, a.v, /*neg_b=*/false, b.v,
        /*c_mod=*/(short)0, c, /*reuse_a=*/false, /*reuse_b=*/false);
}

// float -> bf16 bits, round-to-nearest-even
__device__ inline unsigned short f2bf(float f) {
    unsigned int u = __float_as_uint(f);
    u += 0x7FFFu + ((u >> 16) & 1u);
    return (unsigned short)(u >> 16);
}
__device__ inline float bf2f(unsigned short s) {
    return __uint_as_float(((unsigned int)s) << 16);
}
__device__ inline unsigned int pk2(float lo, float hi) {
    return (unsigned int)f2bf(lo) | ((unsigned int)f2bf(hi) << 16);
}

// ---------------- TDM availability ----------------
#if defined(__has_builtin)
#  if __has_builtin(__builtin_amdgcn_tensor_load_to_lds)
#    define HAVE_TDM 1
#  endif
#endif
#ifndef HAVE_TDM
#  define HAVE_TDM 0
#endif

#if HAVE_TDM
// DMA a [rows x 128] bf16 tile (row stride 128 elems) from global into LDS,
// with HW row padding of 4 DWORDs -> LDS pitch 136 bf16 elements.
__device__ inline void tdm_load_tile_k(unsigned int ldsAddr,
                                       const unsigned short* gsrc, int rows) {
    unsigned long long ga = (unsigned long long)(const void*)gsrc;
    u32x4 g0 = { 1u,                                   // count=1, user mode
                 ldsAddr,                              // lds_addr [63:32]
                 (unsigned int)ga,                     // global_addr lo
                 ((unsigned int)(ga >> 32) & 0x01FFFFFFu) | 0x80000000u }; // type=2
    i32x8 g1 = { (int)0x07510000u,        // data_size=2B, pad_en, interval=64DW, amount=4DW
                 (int)(128u << 16),       // tensor_dim0 = 128 (low16 in [31:16])
                 (int)(2048u << 16),      // tensor_dim1 = 2048
                 (int)(128u << 16),       // tile_dim0 = 128
                 rows,                    // tile_dim1 = rows
                 128,                     // tensor_dim0_stride = 128
                 0, 0 };
    i32x4 z4 = { 0, 0, 0, 0 };
#if __clang_major__ >= 23
    i32x8 z8 = { 0, 0, 0, 0, 0, 0, 0, 0 };
    __builtin_amdgcn_tensor_load_to_lds(g0, g1, z4, z4, z8, 0);
#else
    __builtin_amdgcn_tensor_load_to_lds(g0, g1, z4, z4, 0);
#endif
}
__device__ inline void tdm_wait0() {
#if __has_builtin(__builtin_amdgcn_s_wait_tensorcnt)
    __builtin_amdgcn_s_wait_tensorcnt((short)0);
#else
    asm volatile("s_wait_tensorcnt 0x0" ::: "memory");
#endif
}
#endif

// ============================================================
// Kernel 1: QKV GEMM  (M=4096, K=2048, N=6144), bf16 WMMA.
// Block 256 = 8 waves; tile 128(M) x 128(N) x 32(K);
// wave tile 32x64 = 8 WMMAs per K-step (A reused 4x, B 2x).
// Epilogue: +bias, scatter to q/k/v [B,H,L,HD] bf16 buffers.
// ============================================================
__global__ __launch_bounds__(256) void qkv_gemm(
    const float* __restrict__ x, const float* __restrict__ W,
    const float* __restrict__ bias,
    unsigned short* __restrict__ qb, unsigned short* __restrict__ kbuf,
    unsigned short* __restrict__ vb)
{
    __shared__ __align__(16) unsigned short As[128][36];  // [m][k] bf16
    __shared__ __align__(16) unsigned short Bs[128][36];  // [n][k] bf16 (transposed)

    const int tid  = threadIdx.x;
    const int lane = tid & 31, wave = tid >> 5;
    const int half = lane >> 4, ln = lane & 15;
    const int wm = wave & 3, wn = wave >> 2;              // 4x2 wave grid
    const int nbase = blockIdx.x * 128, mbase = blockIdx.y * 128;

    const int am  = tid >> 1,  ak0 = (tid & 1) * 16;      // A stage: 16 elems/thread
    const int bk  = tid >> 3,  bn0 = (tid & 7) * 16;      // B stage: 16 elems/thread

    v8f acc[2][4];
    #pragma unroll
    for (int mi = 0; mi < 2; ++mi)
        #pragma unroll
        for (int ni = 0; ni < 4; ++ni) acc[mi][ni] = vzero8();

    for (int kb = 0; kb < D_ / 32; ++kb) {
        // stage A tile (f32 -> bf16), coalesced
        {
            const float* xp = x + (size_t)(mbase + am) * D_ + kb * 32 + ak0;
            unsigned int* arow = (unsigned int*)&As[am][ak0];
            #pragma unroll
            for (int q = 0; q < 4; ++q) {
                float4 a4 = *(const float4*)(xp + q * 4);
                arow[q * 2]     = pk2(a4.x, a4.y);
                arow[q * 2 + 1] = pk2(a4.z, a4.w);
            }
        }
        // stage B tile transposed (coalesced global read, scattered LDS write)
        {
            const float* wp = W + (size_t)(kb * 32 + bk) * ND_ + nbase + bn0;
            #pragma unroll
            for (int q = 0; q < 4; ++q) {
                float4 b4 = *(const float4*)(wp + q * 4);
                Bs[bn0 + q * 4 + 0][bk] = f2bf(b4.x);
                Bs[bn0 + q * 4 + 1][bk] = f2bf(b4.y);
                Bs[bn0 + q * 4 + 2][bk] = f2bf(b4.z);
                Bs[bn0 + q * 4 + 3][bk] = f2bf(b4.w);
            }
        }
        __syncthreads();

        ABFrag a0, a1;
        const unsigned int* ap0 = (const unsigned int*)&As[wm * 32 + ln][0];
        const unsigned int* ap1 = (const unsigned int*)&As[wm * 32 + 16 + ln][0];
        #pragma unroll
        for (int j = 0; j < 8; ++j) {
            const int ai = half * 8 + j;  // A: K=half*16+2j
            a0.u[j] = ap0[ai];
            a1.u[j] = ap1[ai];
        }
        #pragma unroll
        for (int ni = 0; ni < 4; ++ni) {
            ABFrag bf;
            const unsigned int* bp =
                (const unsigned int*)&Bs[wn * 64 + ni * 16 + ln][0];
            #pragma unroll
            for (int j = 0; j < 8; ++j) bf.u[j] = bp[half * 8 + j]; // B: K=half*16+2j
            acc[0][ni] = wmma_bf16(a0, bf, acc[0][ni]);
            acc[1][ni] = wmma_bf16(a1, bf, acc[1][ni]);
        }
        __syncthreads();
    }

    // epilogue: bias + scatter to q/k/v
    #pragma unroll
    for (int mi = 0; mi < 2; ++mi)
        #pragma unroll
        for (int ni = 0; ni < 4; ++ni) {
            const int cg = nbase + wn * 64 + ni * 16 + ln;   // col in [0,6144)
            const float bv = bias[cg];
            const int h = cg / (3 * HD_), w = cg - h * (3 * HD_);
            const int sel = w >> 7, wd = w & (HD_ - 1);
            unsigned short* dst = (sel == 0) ? qb : ((sel == 1) ? kbuf : vb);
            #pragma unroll
            for (int r = 0; r < 8; ++r) {
                const int rg  = mbase + wm * 32 + mi * 16 + r + 8 * half;
                const int bi  = rg >> 11, pos = rg & (L_ - 1);
                dst[(((size_t)(bi * H_ + h) * L_ + pos) << 7) + wd] =
                    f2bf(acc[mi][ni][r] + bv);
            }
        }
}

// ============================================================
// Kernel 2: NeoX rotary on first R dims of q and k (in place).
// ============================================================
__global__ __launch_bounds__(256) void rope_kernel(
    unsigned short* __restrict__ qb, unsigned short* __restrict__ kbuf,
    const float* __restrict__ cosb, const float* __restrict__ sinb)
{
    const int idx = blockIdx.x * 256 + threadIdx.x;       // B*H*L*16
    const int d   = idx & 15;
    const int pos = (idx >> 4) & (L_ - 1);
    const int bh  = idx >> 15;
    if (bh >= B_ * H_) return;
    const float c0 = cosb[pos * R_ + d],      s0 = sinb[pos * R_ + d];
    const float c1 = cosb[pos * R_ + d + 16], s1 = sinb[pos * R_ + d + 16];
    const size_t base = ((size_t)bh * L_ + pos) * HD_;
    {
        float a = bf2f(qb[base + d]), b = bf2f(qb[base + d + 16]);
        qb[base + d]      = f2bf(a * c0 - b * s0);
        qb[base + d + 16] = f2bf(b * c1 + a * s1);
    }
    {
        float a = bf2f(kbuf[base + d]), b = bf2f(kbuf[base + d + 16]);
        kbuf[base + d]      = f2bf(a * c0 - b * s0);
        kbuf[base + d + 16] = f2bf(b * c1 + a * s1);
    }
}

// ============================================================
// Kernel 3: causal flash attention. Block 128 (4 waves), each
// block: one (bh, 64-row q-tile); each wave: 16 q-rows.
// K tile DMA'd to LDS by the Tensor Data Mover (wave 0 issues,
// s_wait_tensorcnt, workgroup barrier releases all waves); V
// staged transposed manually; P re-laid out via per-wave LDS.
// ============================================================
__global__ __launch_bounds__(128) void flash_attn(
    const unsigned short* __restrict__ qb,
    const unsigned short* __restrict__ kbuf,
    const unsigned short* __restrict__ vb,
    unsigned short* __restrict__ ob)
{
    __shared__ __align__(16) unsigned short Klds[64][136];   // [kv][d], TDM-padded pitch
    __shared__ __align__(16) unsigned short Vt[128][68];     // [d][kv]
    __shared__ __align__(16) unsigned short Plds[4][16][68]; // per-wave P

    const int tid = threadIdx.x, lane = tid & 31, wave = tid >> 5;
    const int half = lane >> 4, ln = lane & 15;
    const int qt = blockIdx.x, bh = blockIdx.y;
    const int b = bh >> 4, h = bh & 15;
    const int qbase = qt * 64;
    const int qrow  = qbase + wave * 16 + ln;     // A-fragment row (M = lane&15)

    // preload Q fragments for d = 0..127 (4 k-steps of 32)
    const unsigned short* qrp = qb + ((size_t)bh * L_ + qrow) * HD_;
    ABFrag qf[4];
    #pragma unroll
    for (int ks = 0; ks < 4; ++ks)
        #pragma unroll
        for (int j = 0; j < 8; ++j)
            qf[ks].u[j] = *(const unsigned int*)
                (qrp + ks * 32 + half * 16 + j * 2);

    float mi[8], li[8];
    v8f O[8];
    #pragma unroll
    for (int r = 0; r < 8; ++r) { mi[r] = -1e30f; li[r] = 0.f; }
    #pragma unroll
    for (int dt = 0; dt < 8; ++dt) O[dt] = vzero8();

    const int skv = tid >> 1, sd0 = (tid & 1) * 64;   // V staging map: 2 thr/row
    const unsigned short* kbhp = kbuf + (size_t)bh * L_ * HD_;
    const unsigned short* vbhp = vb   + (size_t)bh * L_ * HD_;
#if HAVE_TDM
    const unsigned int kldsAddr =
        (unsigned int)(unsigned long long)(void*)&Klds[0][0];
#endif

    for (int nt = 0; nt <= qt; ++nt) {
        const int kvb = nt * 64;
        // ---- stage K: TDM DMA (row pitch padded 128->136 elems) ----
#if HAVE_TDM
        if (wave == 0) {
            tdm_load_tile_k(kldsAddr, kbhp + (size_t)kvb * HD_, 64);
        }
#else
        {
            const unsigned short* kr = kbhp + (size_t)(kvb + skv) * HD_ + sd0;
            unsigned int* krow = (unsigned int*)&Klds[skv][sd0];
            #pragma unroll
            for (int i = 0; i < 4; ++i) {
                uint4 t4 = *(const uint4*)(kr + i * 16);
                krow[i * 4 + 0] = t4.x; krow[i * 4 + 1] = t4.y;
                krow[i * 4 + 2] = t4.z; krow[i * 4 + 3] = t4.w;
            }
        }
#endif
        // ---- stage V transposed (manual; TDM cannot transpose) ----
        {
            const unsigned short* vr = vbhp + (size_t)(kvb + skv) * HD_ + sd0;
            #pragma unroll
            for (int i = 0; i < 8; ++i) {
                uint4 w4 = *(const uint4*)(vr + i * 8);
                const unsigned short* ws = (const unsigned short*)&w4;
                #pragma unroll
                for (int e = 0; e < 8; ++e) Vt[sd0 + i * 8 + e][skv] = ws[e];
            }
        }
#if HAVE_TDM
        if (wave == 0) tdm_wait0();
#endif
        __syncthreads();

        // ---- S = Q K^T (4 col-tiles x 4 k-steps) ----
        v8f S[4];
        #pragma unroll
        for (int j4 = 0; j4 < 4; ++j4) {
            v8f s = vzero8();
            #pragma unroll
            for (int ks = 0; ks < 4; ++ks) {
                ABFrag bf;
                const unsigned short* krow = &Klds[j4 * 16 + ln][ks * 32];
                #pragma unroll
                for (int j = 0; j < 8; ++j)
                    bf.u[j] = *(const unsigned int*)(krow + half * 16 + j * 2);
                s = wmma_bf16(qf[ks], bf, s);
            }
            S[j4] = s;
        }
        // scale + causal mask (diagonal tile only)
        const bool diag = (nt == qt);
        #pragma unroll
        for (int j4 = 0; j4 < 4; ++j4)
            #pragma unroll
            for (int r = 0; r < 8; ++r) {
                float sv = S[j4][r] * SCALE_;
                if (diag) {
                    const int kvc = kvb + j4 * 16 + ln;
                    const int rr  = qbase + wave * 16 + r + 8 * half;
                    if (kvc > rr) sv = -1e30f;
                }
                S[j4][r] = sv;
            }
        // ---- online softmax (rows live on 16-lane half-groups) ----
        float mnew[8], al[8];
        #pragma unroll
        for (int r = 0; r < 8; ++r) {
            float v = fmaxf(fmaxf(S[0][r], S[1][r]), fmaxf(S[2][r], S[3][r]));
            #pragma unroll
            for (int mk = 1; mk < 16; mk <<= 1) v = fmaxf(v, __shfl_xor(v, mk, 32));
            mnew[r] = fmaxf(mi[r], v);
            al[r]   = __expf(mi[r] - mnew[r]);
            mi[r]   = mnew[r];
        }
        #pragma unroll
        for (int j4 = 0; j4 < 4; ++j4)
            #pragma unroll
            for (int r = 0; r < 8; ++r)
                S[j4][r] = __expf(S[j4][r] - mnew[r]);
        #pragma unroll
        for (int r = 0; r < 8; ++r) {
            float rs = S[0][r] + S[1][r] + S[2][r] + S[3][r];
            #pragma unroll
            for (int mk = 1; mk < 16; mk <<= 1) rs += __shfl_xor(rs, mk, 32);
            li[r] = li[r] * al[r] + rs;
            #pragma unroll
            for (int dt = 0; dt < 8; ++dt) O[dt][r] *= al[r];
        }
        // ---- P (D-tile layout) -> LDS -> A-fragment layout ----
        #pragma unroll
        for (int j4 = 0; j4 < 4; ++j4)
            #pragma unroll
            for (int r = 0; r < 8; ++r)
                Plds[wave][r + 8 * half][j4 * 16 + ln] = f2bf(S[j4][r]);
        __syncthreads();
        // ---- O += P V ----
        #pragma unroll
        for (int kc = 0; kc < 2; ++kc) {
            ABFrag pf;
            const unsigned short* prow = &Plds[wave][ln][kc * 32];
            #pragma unroll
            for (int j = 0; j < 8; ++j)
                pf.u[j] = *(const unsigned int*)
                    (prow + half * 16 + j * 2);
            #pragma unroll
            for (int dt = 0; dt < 8; ++dt) {
                ABFrag vf;
                const unsigned short* vrow = &Vt[dt * 16 + ln][kc * 32];
                #pragma unroll
                for (int j = 0; j < 8; ++j)
                    vf.u[j] = *(const unsigned int*)(vrow + half * 16 + j * 2);
                O[dt] = wmma_bf16(pf, vf, O[dt]);
            }
        }
        __syncthreads();
    }

    // epilogue: normalize, write [B, L, H*HD] bf16
    #pragma unroll
    for (int r = 0; r < 8; ++r) {
        const int rr = qbase + wave * 16 + r + 8 * half;
        const float inv = 1.0f / li[r];
        unsigned short* op = ob + ((size_t)(b * L_ + rr)) * D_ + h * HD_;
        #pragma unroll
        for (int dt = 0; dt < 8; ++dt)
            op[dt * 16 + ln] = f2bf(O[dt][r] * inv);
    }
}

// ============================================================
// Kernel 4: output projection  (M=4096, K=2048, N=2048),
// bf16 A from flash output, f32 result + bias into d_out.
// Same 128x128x32 tiling as kernel 1.
// ============================================================
__global__ __launch_bounds__(256) void out_gemm(
    const unsigned short* __restrict__ A, const float* __restrict__ W,
    const float* __restrict__ bias, float* __restrict__ out)
{
    __shared__ __align__(16) unsigned short As[128][36];
    __shared__ __align__(16) unsigned short Bs[128][36];

    const int tid  = threadIdx.x;
    const int lane = tid & 31, wave = tid >> 5;
    const int half = lane >> 4, ln = lane & 15;
    const int wm = wave & 3, wn = wave >> 2;
    const int nbase = blockIdx.x * 128, mbase = blockIdx.y * 128;

    const int am = tid >> 1, ak0 = (tid & 1) * 16;
    const int bk = tid >> 3, bn0 = (tid & 7) * 16;

    v8f acc[2][4];
    #pragma unroll
    for (int mi = 0; mi < 2; ++mi)
        #pragma unroll
        for (int ni = 0; ni < 4; ++ni) acc[mi][ni] = vzero8();

    for (int kb = 0; kb < D_ / 32; ++kb) {
        {
            const unsigned short* ap = A + (size_t)(mbase + am) * D_ + kb * 32 + ak0;
            unsigned int* arow = (unsigned int*)&As[am][ak0];
            #pragma unroll
            for (int q = 0; q < 2; ++q) {
                uint4 av = *(const uint4*)(ap + q * 8);   // already bf16
                arow[q * 4 + 0] = av.x; arow[q * 4 + 1] = av.y;
                arow[q * 4 + 2] = av.z; arow[q * 4 + 3] = av.w;
            }
        }
        {
            const float* wp = W + (size_t)(kb * 32 + bk) * D_ + nbase + bn0;
            #pragma unroll
            for (int q = 0; q < 4; ++q) {
                float4 b4 = *(const float4*)(wp + q * 4);
                Bs[bn0 + q * 4 + 0][bk] = f2bf(b4.x);
                Bs[bn0 + q * 4 + 1][bk] = f2bf(b4.y);
                Bs[bn0 + q * 4 + 2][bk] = f2bf(b4.z);
                Bs[bn0 + q * 4 + 3][bk] = f2bf(b4.w);
            }
        }
        __syncthreads();

        ABFrag a0, a1;
        const unsigned int* ap0 = (const unsigned int*)&As[wm * 32 + ln][0];
        const unsigned int* ap1 = (const unsigned int*)&As[wm * 32 + 16 + ln][0];
        #pragma unroll
        for (int j = 0; j < 8; ++j) {
            const int ai = half * 8 + j;
            a0.u[j] = ap0[ai];
            a1.u[j] = ap1[ai];
        }
        #pragma unroll
        for (int ni = 0; ni < 4; ++ni) {
            ABFrag bf;
            const unsigned int* bp =
                (const unsigned int*)&Bs[wn * 64 + ni * 16 + ln][0];
            #pragma unroll
            for (int j = 0; j < 8; ++j) bf.u[j] = bp[half * 8 + j];
            acc[0][ni] = wmma_bf16(a0, bf, acc[0][ni]);
            acc[1][ni] = wmma_bf16(a1, bf, acc[1][ni]);
        }
        __syncthreads();
    }

    #pragma unroll
    for (int mi = 0; mi < 2; ++mi)
        #pragma unroll
        for (int ni = 0; ni < 4; ++ni) {
            const int cg = nbase + wn * 64 + ni * 16 + ln;
            const float bv = bias[cg];
            #pragma unroll
            for (int r = 0; r < 8; ++r) {
                const int rg = mbase + wm * 32 + mi * 16 + r + 8 * half;
                out[(size_t)rg * D_ + cg] = acc[mi][ni][r] + bv;
            }
        }
}

// ============================================================
extern "C" void kernel_launch(void* const* d_in, const int* in_sizes, int n_in,
                              void* d_out, int out_size, void* d_ws, size_t ws_size,
                              hipStream_t stream)
{
    (void)in_sizes; (void)n_in; (void)out_size; (void)ws_size;
    const float* x    = (const float*)d_in[0];
    const float* cosb = (const float*)d_in[1];
    const float* sinb = (const float*)d_in[2];
    const float* Wqkv = (const float*)d_in[3];
    const float* bqkv = (const float*)d_in[4];
    const float* Wd   = (const float*)d_in[5];
    const float* bd   = (const float*)d_in[6];

    // workspace: 4 bf16 buffers of B*H*L*HD elements (16 MB each, 64 MB total)
    unsigned short* qb = (unsigned short*)d_ws;
    unsigned short* kb = qb + ELEMS_;
    unsigned short* vb = kb + ELEMS_;
    unsigned short* ob = vb + ELEMS_;

    qkv_gemm<<<dim3(ND_ / 128, (B_ * L_) / 128), 256, 0, stream>>>(
        x, Wqkv, bqkv, qb, kb, vb);
    rope_kernel<<<(B_ * H_ * L_ * 16) / 256, 256, 0, stream>>>(
        qb, kb, cosb, sinb);
    flash_attn<<<dim3(L_ / 64, B_ * H_), 128, 0, stream>>>(qb, kb, vb, ob);
    out_gemm<<<dim3(D_ / 128, (B_ * L_) / 128), 256, 0, stream>>>(
        ob, Wd, bd, (float*)d_out);
}
